// RSSM_35476429865163
// MI455X (gfx1250) — matrix operand order repaired and checked by reading, not compiled, not run on hardware
//
#include <hip/hip_runtime.h>

// ---------------------------------------------------------------------------
// RSSM GRU scan, MI455X (gfx1250, wave32, WMMA).
// bf16 WMMA GEMMs with fp32 accumulate; weights converted to bf16 once
// (~63MB -> L2-resident on the 192MB L2). Register-blocked 4x2 wave tiles
// (8 WMMAs per 12 b128 loads) + K-split-4 with LDS reduction for GEMM1 to
// keep ~512 waves in flight on the latency-critical sequential scan.
// ---------------------------------------------------------------------------

typedef __attribute__((ext_vector_type(16))) __bf16 v16bf;
typedef __attribute__((ext_vector_type(8)))  __bf16 v8bf;
typedef __attribute__((ext_vector_type(8)))  float  v8f;

#define B_  64
#define T_  64
#define E_  2048
#define D_  4096
#define G_  8
#define H3_ (3 * D_)       // 12288
#define GK_ (D_ / G_)      // 512
#define GN_ (H3_ / G_)     // 1536
#define EPS_ 1e-4f

// ---------------- elementwise converts ----------------
__global__ __launch_bounds__(256)
void k_f32_to_bf16(const float* __restrict__ src, __bf16* __restrict__ dst, int n) {
    for (int i = blockIdx.x * 256 + threadIdx.x; i < n; i += gridDim.x * 256)
        dst[i] = (__bf16)src[i];
}

__global__ __launch_bounds__(256)
void k_init_deter(const float* __restrict__ src, float* __restrict__ df,
                  __bf16* __restrict__ db) {
    int i = blockIdx.x * 256 + threadIdx.x;   // grid sized exactly B*D/256
    float v = src[i];
    df[i] = v;
    db[i] = (__bf16)v;
}

// ---------------- helpers ----------------
__device__ __forceinline__ v16bf load_a_frag(const __bf16* __restrict__ row,
                                             int kb, int half) {
    // CDNA5 16-bit A layout: lane<16 holds K {kb..kb+7, kb+16..kb+23},
    // lane>=16 holds K {kb+8..kb+15, kb+24..kb+31}; two 16B contiguous loads.
    v8bf lo = *(const v8bf*)(row + kb + half * 8);
    v8bf hi = *(const v8bf*)(row + kb + 16 + half * 8);
    return __builtin_shufflevector(lo, hi, 0, 1, 2, 3, 4, 5, 6, 7,
                                           8, 9, 10, 11, 12, 13, 14, 15);
}

__device__ __forceinline__ float sigm(float x) { return 1.0f / (1.0f + __expf(-x)); }

// ---------------- GEMM1: z1[64,4096] = concat(deter,x_t)[64,6144] @ W_in + b_in
// Block = 4 waves, 32-column strip. Wave tile: M=64 (4 m-tiles) x N=32
// (2 n-tiles), one K-chunk of 1536 per wave; partials reduced through LDS.
// grid.x = N/32 = 128.
__global__ __launch_bounds__(128)
void k_gemm1(const __bf16* __restrict__ deter_bf,
             const __bf16* __restrict__ xbf, int t,
             const __bf16* __restrict__ Wbf,
             const float*  __restrict__ b_in,
             float* __restrict__ z1) {
    const int lane = threadIdx.x & 31;
    const int wave = threadIdx.x >> 5;     // = K-chunk index (0..3)
    const int half = lane >> 4;
    const int l16  = lane & 15;
    const int n0   = blockIdx.x * 32;      // block's 32-column strip

    const int k0 = wave * 1536;
    const int k1 = k0 + 1536;
    const int kd_end   = (k1 < D_) ? k1 : D_;   // deter part end
    const int kx_start = (k0 > D_) ? k0 : D_;   // x part start

    v8f acc[8] = {};                       // [mt*2 + nt]

    const __bf16* __restrict__ Bcol = Wbf + n0;

    // A row pointers: 4 m-tiles (rows l16, l16+16, l16+32, l16+48)
    const __bf16* __restrict__ arow[4];
    const __bf16* __restrict__ xrow[4];
#pragma unroll
    for (int mt = 0; mt < 4; ++mt) {
        arow[mt] = deter_bf + (size_t)(mt * 16 + l16) * D_;
        xrow[mt] = xbf + ((size_t)(mt * 16 + l16) * T_ + t) * E_;
    }

    // K-phase 1: deter part of H
#pragma unroll 1
    for (int k = k0; k < kd_end; k += 32) {
        v16bf a[4], b[2];
#pragma unroll
        for (int mt = 0; mt < 4; ++mt) a[mt] = load_a_frag(arow[mt], k, half);
#pragma unroll
        for (int nt = 0; nt < 2; ++nt)
#pragma unroll
            for (int s = 0; s < 16; ++s)
                b[nt][s] = Bcol[(size_t)(k + (s < 8 ? half * 8 + s : 16 + half * 8 + s - 8)) * D_ + nt * 16 + l16];
        __builtin_prefetch(Bcol + (size_t)(k + 32 + lane) * D_, 0, 0);
#pragma unroll
        for (int mt = 0; mt < 4; ++mt)
#pragma unroll
            for (int nt = 0; nt < 2; ++nt)
                acc[mt * 2 + nt] = __builtin_amdgcn_wmma_f32_16x16x32_bf16(
                    false, a[mt], false, b[nt], (short)0, acc[mt * 2 + nt],
                    false, false);
    }
    // K-phase 2: x_t part of H
#pragma unroll 1
    for (int k = kx_start; k < k1; k += 32) {
        v16bf a[4], b[2];
#pragma unroll
        for (int mt = 0; mt < 4; ++mt) a[mt] = load_a_frag(xrow[mt], k - D_, half);
#pragma unroll
        for (int nt = 0; nt < 2; ++nt)
#pragma unroll
            for (int s = 0; s < 16; ++s)
                b[nt][s] = Bcol[(size_t)(k + (s < 8 ? half * 8 + s : 16 + half * 8 + s - 8)) * D_ + nt * 16 + l16];
        if (k + 32 < k1)
            __builtin_prefetch(Bcol + (size_t)(k + 32 + lane) * D_, 0, 0);
#pragma unroll
        for (int mt = 0; mt < 4; ++mt)
#pragma unroll
            for (int nt = 0; nt < 2; ++nt)
                acc[mt * 2 + nt] = __builtin_amdgcn_wmma_f32_16x16x32_bf16(
                    false, a[mt], false, b[nt], (short)0, acc[mt * 2 + nt],
                    false, false);
    }

    // cross-wave (K-chunk) reduction through LDS: [4 waves][64 rows][32 cols]
    __shared__ float lred[4 * 64 * 32];
    float* __restrict__ myl = lred + wave * (64 * 32);
#pragma unroll
    for (int mt = 0; mt < 4; ++mt)
#pragma unroll
        for (int nt = 0; nt < 2; ++nt) {
            v8f a8 = acc[mt * 2 + nt];
#pragma unroll
            for (int r = 0; r < 8; ++r) {
                int m  = mt * 16 + r + half * 8;
                int nl = nt * 16 + l16;
                myl[m * 32 + nl] = a8[r];
            }
        }
    __syncthreads();
    for (int i = threadIdx.x; i < 64 * 32; i += 128) {
        int m  = i >> 5;
        int nl = i & 31;
        float s = lred[i] + lred[2048 + i] + lred[4096 + i] + lred[6144 + i];
        int n = n0 + nl;
        z1[(size_t)m * D_ + n] = s + b_in[n];
    }
}

// ---------------- full-row RMS + SiLU -> y (bf16). one block per batch row.
__global__ __launch_bounds__(256)
void k_rms_silu(const float* __restrict__ z1, const float* __restrict__ scale_in,
                __bf16* __restrict__ ybf) {
    __shared__ float red[256];
    const int b = blockIdx.x;
    const float* __restrict__ row = z1 + (size_t)b * D_;
    float s = 0.0f;
    for (int i = threadIdx.x; i < D_; i += 256) { float v = row[i]; s += v * v; }
    red[threadIdx.x] = s;
    __syncthreads();
    for (int off = 128; off > 0; off >>= 1) {
        if (threadIdx.x < off) red[threadIdx.x] += red[threadIdx.x + off];
        __syncthreads();
    }
    const float rstd = rsqrtf(red[0] * (1.0f / D_) + EPS_);
    for (int i = threadIdx.x; i < D_; i += 256) {
        float v = row[i] * rstd * scale_in[i];
        ybf[(size_t)b * D_ + i] = (__bf16)(v * sigm(v));      // silu
    }
}

// ---------------- GEMM2: block-diagonal y[64,8x512] @ W_blk[8,512,1536] + b_blk
// Block = 4 waves covering 128 cols of one group; wave tile M=64 x N=32,
// full K=512. grid (GN/128=12, G=8).
__global__ __launch_bounds__(128)
void k_gemm2(const __bf16* __restrict__ ybf, const __bf16* __restrict__ Wblk,
             const float* __restrict__ b_blk, float* __restrict__ z2) {
    const int lane = threadIdx.x & 31;
    const int wave = threadIdx.x >> 5;
    const int half = lane >> 4;
    const int l16  = lane & 15;
    const int g    = blockIdx.y;
    const int n0   = blockIdx.x * 128 + wave * 32;  // 0..1535 within group

    v8f acc[8] = {};                                // [mt*2 + nt]

    const __bf16* __restrict__ Bbase = Wblk + (size_t)g * GK_ * GN_ + n0;
    const __bf16* __restrict__ arow[4];
#pragma unroll
    for (int mt = 0; mt < 4; ++mt)
        arow[mt] = ybf + (size_t)(mt * 16 + l16) * D_ + g * GK_;

#pragma unroll 1
    for (int k = 0; k < GK_; k += 32) {
        v16bf a[4], b[2];
#pragma unroll
        for (int mt = 0; mt < 4; ++mt) a[mt] = load_a_frag(arow[mt], k, half);
#pragma unroll
        for (int nt = 0; nt < 2; ++nt)
#pragma unroll
            for (int s = 0; s < 16; ++s)
                b[nt][s] = Bbase[(size_t)(k + (s < 8 ? half * 8 + s : 16 + half * 8 + s - 8)) * GN_ + nt * 16 + l16];
        if (k + 32 < GK_)
            __builtin_prefetch(Bbase + (size_t)(k + 32 + lane) * GN_, 0, 0);
#pragma unroll
        for (int mt = 0; mt < 4; ++mt)
#pragma unroll
            for (int nt = 0; nt < 2; ++nt)
                acc[mt * 2 + nt] = __builtin_amdgcn_wmma_f32_16x16x32_bf16(
                    false, a[mt], false, b[nt], (short)0, acc[mt * 2 + nt],
                    false, false);
    }

#pragma unroll
    for (int mt = 0; mt < 4; ++mt)
#pragma unroll
        for (int nt = 0; nt < 2; ++nt) {
            v8f a8 = acc[mt * 2 + nt];
            const int col = g * GN_ + n0 + nt * 16 + l16;
            const float bias = b_blk[col];
#pragma unroll
            for (int r = 0; r < 8; ++r) {
                int m = mt * 16 + r + half * 8;
                z2[(size_t)m * H3_ + col] = a8[r] + bias;
            }
        }
}

// ---------------- per-block RMS (1536-wide chunks), in place. block per (b,g).
__global__ __launch_bounds__(256)
void k_blockrms(float* __restrict__ z2, const float* __restrict__ scale_blk) {
    __shared__ float red[256];
    const int b = blockIdx.x >> 3;          // /G_
    const int g = blockIdx.x & 7;           // %G_
    float* __restrict__ chunk = z2 + (size_t)b * H3_ + g * GN_;
    const float* __restrict__ sc = scale_blk + g * GN_;
    float s = 0.0f;
    for (int i = threadIdx.x; i < GN_; i += 256) { float v = chunk[i]; s += v * v; }
    red[threadIdx.x] = s;
    __syncthreads();
    for (int off = 128; off > 0; off >>= 1) {
        if (threadIdx.x < off) red[threadIdx.x] += red[threadIdx.x + off];
        __syncthreads();
    }
    const float rstd = rsqrtf(red[0] * (1.0f / GN_) + EPS_);
    for (int i = threadIdx.x; i < GN_; i += 256)
        chunk[i] = chunk[i] * rstd * sc[i];
}

// ---------------- gates + deter update; writes out[:,t,:] and next-step bf16
__global__ __launch_bounds__(256)
void k_gate(const float* __restrict__ z2, float* __restrict__ deter_f,
            __bf16* __restrict__ deter_bf, float* __restrict__ out, int t) {
    const int idx = blockIdx.x * 256 + threadIdx.x;  // grid = B*D/256 exactly
    const int b = idx >> 12;            // /D_
    const int d = idx & (D_ - 1);       // %D_
    const float* __restrict__ row = z2 + (size_t)b * H3_;
    const float r = row[d];
    const float c = row[D_ + d];
    const float u = row[2 * D_ + d];
    const float reset  = sigm(r);
    const float cand   = tanhf(reset * c);
    const float update = sigm(u - 1.0f);
    const float dn = update * cand + (1.0f - update) * deter_f[idx];
    deter_f[idx]  = dn;
    deter_bf[idx] = (__bf16)dn;
    out[((size_t)b * T_ + t) * D_ + d] = dn;
}

// ---------------------------------------------------------------------------
extern "C" void kernel_launch(void* const* d_in, const int* in_sizes, int n_in,
                              void* d_out, int out_size, void* d_ws, size_t ws_size,
                              hipStream_t stream) {
    (void)in_sizes; (void)n_in; (void)out_size; (void)ws_size;

    const float* x         = (const float*)d_in[0];   // [B,T,E]
    const float* deter0    = (const float*)d_in[1];   // [B,D]
    const float* W_in      = (const float*)d_in[2];   // [D+E, D]
    const float* b_in      = (const float*)d_in[3];   // [D]
    const float* scale_in  = (const float*)d_in[4];   // [D]
    const float* W_blk     = (const float*)d_in[5];   // [G, D/G, 3D/G]
    const float* b_blk     = (const float*)d_in[6];   // [3D]
    const float* scale_blk = (const float*)d_in[7];   // [G, 3D/G]
    float* out = (float*)d_out;                       // [B,T,D] fp32

    // workspace carve-out (256B aligned slices)
    char* p = (char*)d_ws;
    auto carve = [&](size_t bytes) {
        char* r = p;
        p += (bytes + 255) & ~(size_t)255;
        return r;
    };
    const int nWin  = (D_ + E_) * D_;      // 25,165,824
    const int nWblk = G_ * GK_ * GN_;      //  6,291,456
    const int nX    = B_ * T_ * E_;        //  8,388,608
    const int nBD   = B_ * D_;             //    262,144

    __bf16* Wbf      = (__bf16*)carve((size_t)nWin  * 2);
    __bf16* Wblk_bf  = (__bf16*)carve((size_t)nWblk * 2);
    __bf16* xbf      = (__bf16*)carve((size_t)nX    * 2);
    float*  deter_f  = (float*) carve((size_t)nBD   * 4);
    __bf16* deter_bf = (__bf16*)carve((size_t)nBD   * 2);
    __bf16* ybf      = (__bf16*)carve((size_t)nBD   * 2);
    float*  z1       = (float*) carve((size_t)nBD   * 4);
    float*  z2       = (float*) carve((size_t)(B_ * H3_) * 4);

    // one-time (per launch) bf16 conversions — weights then stay L2-resident
    k_f32_to_bf16<<<4096, 256, 0, stream>>>(W_in,  Wbf,     nWin);
    k_f32_to_bf16<<<2048, 256, 0, stream>>>(W_blk, Wblk_bf, nWblk);
    k_f32_to_bf16<<<2048, 256, 0, stream>>>(x,     xbf,     nX);
    k_init_deter<<<nBD / 256, 256, 0, stream>>>(deter0, deter_f, deter_bf);

    for (int t = 0; t < T_; ++t) {
        k_gemm1<<<D_ / 32, 128, 0, stream>>>(deter_bf, xbf, t, Wbf, b_in, z1);
        k_rms_silu<<<B_, 256, 0, stream>>>(z1, scale_in, ybf);
        k_gemm2<<<dim3(GN_ / 128, G_), 128, 0, stream>>>(ybf, Wblk_bf, b_blk, z2);
        k_blockrms<<<B_ * G_, 256, 0, stream>>>(z2, scale_blk);
        k_gate<<<nBD / 256, 256, 0, stream>>>(z2, deter_f, deter_bf, out, t);
    }
}
